// SharedBarycenterLoss_72679436583204
// MI455X (gfx1250) — matrix operand order repaired and checked by reading, not compiled, not run on hardware
//
#include <hip/hip_runtime.h>
#include <hip/hip_bf16.h>
#include <math.h>

// ---------------------------------------------------------------------------
// SharedBarycenterLoss for MI455X (gfx1250, wave32, WMMA).
//
// V diag(1/(1+lam*w)) V^T bary == (I + lam*L)^{-1} bary, so the eigh is
// replaced by CG whose inner op is a dense (ns x ns)@(ns x 256) GEMM done
// with v_wmma_f32_16x16x32_bf16 (bf16 operands, f32 accumulate).
//
// GEMM operands are pre-packed bf16 (halves the CG streaming traffic of the
// dense operator against 23.3 TB/s HBM / 192 MB L2). For BT=0 GEMMs the
// 32x16 B tile is staged through LDS in WMMA fragment order (b32 global
// loads -> ds_store_b16, ds_load_b128 reads, double-buffered, one barrier
// per K-step) and shared by the 8 waves of a block.
// ---------------------------------------------------------------------------

#define REG_EPS   0.05f
#define REG_M     0.5f
#define N_SINK    100
#define K_NN      10
#define LAM_EIG   0.5f
#define EPSV      1e-16f
#define N_CG      32

typedef __attribute__((ext_vector_type(16))) __bf16    bf16x16;
typedef __attribute__((ext_vector_type(8)))  float     f32x8;
typedef __attribute__((ext_vector_type(4)))  unsigned  u32x4;

// ------------------------------ elementwise --------------------------------

__global__ void k_fill(float* p, long n, float val) {
  long i = (long)blockIdx.x * blockDim.x + threadIdx.x;
  long stride = (long)gridDim.x * blockDim.x;
  for (; i < n; i += stride) p[i] = val;
}

__global__ void k_cvt_bf16(const float* __restrict__ src, __bf16* __restrict__ dst,
                           long n) {
  long i = (long)blockIdx.x * blockDim.x + threadIdx.x;
  long stride = (long)gridDim.x * blockDim.x;
  for (; i < n; i += stride) dst[i] = (__bf16)src[i];
}

// K = exp(-C/reg) and KT = K^T (scattered transpose store; one pass over C).
__global__ void k_exp_build(const float* __restrict__ C, float* __restrict__ Km,
                            float* __restrict__ KT, int na, int nb) {
  long n = (long)na * nb;
  long i = (long)blockIdx.x * blockDim.x + threadIdx.x;
  long stride = (long)gridDim.x * blockDim.x;
  for (; i < n; i += stride) {
    int r = (int)(i / nb), c = (int)(i % nb);
    float val = expf(-C[i] * (1.0f / REG_EPS));
    Km[i] = val;
    KT[(size_t)c * na + r] = val;
  }
}

// P = u[i] * K[i,j] * v[j]
__global__ void k_build_P(const float* __restrict__ Km, const float* __restrict__ u,
                          const float* __restrict__ v, float* __restrict__ P,
                          int na, int nb) {
  long n = (long)na * nb;
  long i = (long)blockIdx.x * blockDim.x + threadIdx.x;
  long stride = (long)gridDim.x * blockDim.x;
  for (; i < n; i += stride) {
    int r = (int)(i / nb), c = (int)(i % nb);
    P[i] = u[r] * Km[i] * v[c];
  }
}

__global__ void k_scale_rows(float* __restrict__ X, const float* __restrict__ dnm,
                             int rows, int cols) {
  long n = (long)rows * cols;
  long i = (long)blockIdx.x * blockDim.x + threadIdx.x;
  long stride = (long)gridDim.x * blockDim.x;
  for (; i < n; i += stride) X[i] /= dnm[i / cols];
}

__global__ void k_cg_init(float* __restrict__ X, float* __restrict__ R,
                          float* __restrict__ Dir, const float* __restrict__ bary,
                          long n) {
  long i = (long)blockIdx.x * blockDim.x + threadIdx.x;
  long stride = (long)gridDim.x * blockDim.x;
  for (; i < n; i += stride) { float b = bary[i]; X[i] = 0.0f; R[i] = b; Dir[i] = b; }
}

__global__ void k_cg_xr(float* __restrict__ X, float* __restrict__ R,
                        const float* __restrict__ Dir, const float* __restrict__ AP,
                        const float* __restrict__ rr, const float* __restrict__ pAp,
                        long n, int cols) {
  long i = (long)blockIdx.x * blockDim.x + threadIdx.x;
  long stride = (long)gridDim.x * blockDim.x;
  for (; i < n; i += stride) {
    int c = (int)(i % cols);
    float alpha = rr[c] / (pAp[c] + 1e-30f);
    X[i] += alpha * Dir[i];
    R[i] -= alpha * AP[i];
  }
}

__global__ void k_cg_dir(float* __restrict__ Dir, const float* __restrict__ R,
                         const float* __restrict__ rrn, const float* __restrict__ rro,
                         long n, int cols) {
  long i = (long)blockIdx.x * blockDim.x + threadIdx.x;
  long stride = (long)gridDim.x * blockDim.x;
  for (; i < n; i += stride) {
    int c = (int)(i % cols);
    float beta = rrn[c] / (rro[c] + 1e-30f);
    Dir[i] = R[i] + beta * Dir[i];
  }
}

// --------------------------- wave-per-row reductions -----------------------

__device__ __forceinline__ float wave_sum(float s) {
#pragma unroll
  for (int off = 16; off > 0; off >>= 1) s += __shfl_down(s, off, 32);
  return s;
}

// y[row] = (marg / (dot(K[row,:], x) + eps)) ^ fi    — one wave32 per row.
__global__ void k_sink(const float* __restrict__ Km, const float* __restrict__ x,
                       float* __restrict__ y, int rows, int cols,
                       float marg, float fi) {
  int gw = (int)((blockIdx.x * blockDim.x + threadIdx.x) >> 5);
  int lane = threadIdx.x & 31;
  if (gw >= rows) return;
  const float* rowp = Km + (size_t)gw * cols;
  float s = 0.0f;
  for (int j = lane; j < cols; j += 32) {
    if (j + 256 < cols) __builtin_prefetch(rowp + j + 256, 0, 1);  // global_prefetch_b8
    s += rowp[j] * x[j];
  }
  s = wave_sum(s);
  if (lane == 0) y[gw] = powf(marg / (s + EPSV), fi);
}

__global__ void k_rowsum(const float* __restrict__ P, float* __restrict__ y,
                         int rows, int cols) {
  int gw = (int)((blockIdx.x * blockDim.x + threadIdx.x) >> 5);
  int lane = threadIdx.x & 31;
  if (gw >= rows) return;
  const float* rowp = P + (size_t)gw * cols;
  float s = 0.0f;
  for (int j = lane; j < cols; j += 32) s += rowp[j];
  s = wave_sum(s);
  if (lane == 0) y[gw] = s + EPSV;
}

__global__ void k_colsum(const float* __restrict__ P, float* __restrict__ y,
                         int rows, int cols) {
  int gw = (int)((blockIdx.x * blockDim.x + threadIdx.x) >> 5);
  int lane = threadIdx.x & 31;
  if (gw >= cols) return;
  float s = 0.0f;
  for (int i = lane; i < rows; i += 32) s += P[(size_t)i * cols + gw];
  s = wave_sum(s);
  if (lane == 0) y[gw] = s + EPSV;
}

__global__ void k_rownormsq(const float* __restrict__ z, float* __restrict__ y,
                            int rows, int cols) {
  int gw = (int)((blockIdx.x * blockDim.x + threadIdx.x) >> 5);
  int lane = threadIdx.x & 31;
  if (gw >= rows) return;
  const float* rowp = z + (size_t)gw * cols;
  float s = 0.0f;
  for (int j = lane; j < cols; j += 32) { float t = rowp[j]; s += t * t; }
  s = wave_sum(s);
  if (lane == 0) y[gw] = s;
}

// out[col] = sum_r X[r,col] * Y[r,col]   — one block per column.
__global__ void k_coldot(const float* __restrict__ X, const float* __restrict__ Y,
                         float* __restrict__ out, int rows, int cols) {
  __shared__ float sm[256];
  int c = blockIdx.x, t = threadIdx.x;
  float acc = 0.0f;
  for (int r = t; r < rows; r += 256) acc += X[(size_t)r * cols + c] * Y[(size_t)r * cols + c];
  sm[t] = acc; __syncthreads();
  for (int off = 128; off > 0; off >>= 1) {
    if (t < off) sm[t] += sm[t + off];
    __syncthreads();
  }
  if (t == 0) out[c] = sm[0];
}

// loss += sum((z - X)^2) * scale
__global__ void k_loss(const float* __restrict__ z, const float* __restrict__ X,
                       long n, float scale, float* __restrict__ out) {
  __shared__ float sm[256];
  long i = (long)blockIdx.x * blockDim.x + threadIdx.x;
  long stride = (long)gridDim.x * blockDim.x;
  float acc = 0.0f;
  for (; i < n; i += stride) { float d = z[i] - X[i]; acc += d * d; }
  int t = threadIdx.x;
  sm[t] = acc; __syncthreads();
  for (int off = 128; off > 0; off >>= 1) {
    if (t < off) sm[t] += sm[t + off];
    __syncthreads();
  }
  if (t == 0) atomicAdd(out, sm[0] * scale);
}

// ------------------------------ kNN / Laplacian ----------------------------

// Row-wise k-NN via iterative block argmin over d2 = sq_i + sq_j - 2*G(i,j).
__global__ void k_knn(const float* __restrict__ G, const float* __restrict__ sq,
                      float* __restrict__ Aadj, int n, int k) {
  __shared__ float sval[256];
  __shared__ int   sidx[256];
  __shared__ int   ssel[16];
  int i = blockIdx.x, t = threadIdx.x;
  float sqi = sq[i];
  for (int p = 0; p < k; ++p) {
    float best = INFINITY; int bj = -1;
    for (int j = t; j < n; j += 256) {
      if (j == i) continue;
      bool taken = false;
      for (int q = 0; q < p; ++q) taken |= (ssel[q] == j);
      if (taken) continue;
      float d2 = sqi + sq[j] - 2.0f * G[(size_t)i * n + j];
      if (d2 < best) { best = d2; bj = j; }
    }
    sval[t] = best; sidx[t] = bj;
    __syncthreads();
    for (int off = 128; off > 0; off >>= 1) {
      if (t < off && sval[t + off] < sval[t]) { sval[t] = sval[t + off]; sidx[t] = sidx[t + off]; }
      __syncthreads();
    }
    if (t == 0) { ssel[p] = sidx[0]; Aadj[(size_t)i * n + sidx[0]] = 1.0f; }
    __syncthreads();
  }
}

// M = I + lam*L with L = D - max(A, A^T); one block per row.
__global__ void k_build_M(const float* __restrict__ Aadj, float* __restrict__ Mout,
                          int n, float lam) {
  __shared__ float sdeg[256];
  int i = blockIdx.x, t = threadIdx.x;
  float deg = 0.0f;
  for (int j = t; j < n; j += 256) {
    float s = fmaxf(Aadj[(size_t)i * n + j], Aadj[(size_t)j * n + i]);
    deg += s;
    Mout[(size_t)i * n + j] = -lam * s;
  }
  sdeg[t] = deg; __syncthreads();
  for (int off = 128; off > 0; off >>= 1) {
    if (t < off) sdeg[t] += sdeg[t + off];
    __syncthreads();
  }
  if (t == 0) Mout[(size_t)i * n + i] = 1.0f + lam * sdeg[0];
}

// ------------------------------- WMMA GEMM ---------------------------------
// Fragment layouts per CDNA5 ISA 7.12.2:
//   A (16x32): lane half h, r=lane&15 -> row M=r; elements = two contiguous
//              8-run K windows {kk+8h..} and {kk+16+8h..}  -> 2x b128 (AT=0).
//   B (32x16): lane half h -> K windows {kk+8h..+7} and {kk+16+8h..+7}, N = r.
//   C/D 16x16: VGPR i -> M=i / M=8+i, N = lane&15.

__device__ __forceinline__ void load_a_frag(const __bf16* __restrict__ A, int lda,
                                            int row, int kk, int half, bf16x16& v,
                                            u32x4* q) {
  const __bf16* ap = A + (size_t)row * lda + kk + (half << 3);
  q[0] = *(const u32x4*)ap;         // K = kk + 8h   .. +7
  q[1] = *(const u32x4*)(ap + 16);  // K = kk+16+8h  .. +7
  (void)v;
}

// ---- BT=0 GEMM with LDS-staged, wave-shared B tile ------------------------
// Block = 8 waves: one 16-wide N-tile, 8 consecutive 16-row M-tiles.
// Requires M % 128 == 0, N % 16 == 0, K % 32 == 0.
// The 32x16 bf16 B tile is staged into LDS pre-swizzled into fragment order
// (lane-major), double-buffered: one __syncthreads per K-step; each lane then
// reads its 32B fragment with 2x ds_load_b128.
template <int AT>
__global__ void k_gemm_sh(const __bf16* __restrict__ A, const __bf16* __restrict__ B,
                          float* __restrict__ Dm, int M, int N, int K,
                          int lda, int ldb, int ldd) {
  __shared__ __attribute__((aligned(32))) __bf16 sB[2][512];
  const int mblocks = M >> 7;                // 128-row blocks
  const int tn  = blockIdx.x / mblocks;
  const int tmb = blockIdx.x % mblocks;
  const int t = threadIdx.x;
  const int wv = t >> 5;
  const int lane = t & 31;
  const int half = lane >> 4, r = lane & 15;
  const int row0 = (tmb << 7) + (wv << 4);
  const int col0 = tn << 4;
  // staging map: elements idx=2t,2t+1 of the 512-elem tile; k=idx>>4, n=idx&15
  const int sk = t >> 3;                     // 0..31
  const int sn = (t << 1) & 15;              // even n
  const int lbase = (sn + ((sk & 8) << 1)) << 4;   // fragment slot (lane-major) * 16
  const int e0 = (sk & 7) | ((sk & 16) >> 1);

  f32x8 acc = {};
  int buf = 0;
  for (int kk = 0; kk < K; kk += 32, buf ^= 1) {
    // cooperative B stage: one b32 (two adjacent-N bf16), two ds_store_b16
    union { unsigned u; __bf16 h[2]; } pk;
    pk.u = *(const unsigned*)(B + (size_t)(kk + sk) * ldb + col0 + sn);
    sB[buf][lbase + e0]      = pk.h[0];      // lane sn+..,   element e0
    sB[buf][lbase + 16 + e0] = pk.h[1];      // lane sn+1+.., element e0
    __syncthreads();

    union { bf16x16 v; u32x4 q[2]; } fa, fb;
    if constexpr (AT == 0) {
      const __bf16* ap = A + (size_t)(row0 + r) * lda + kk + (half << 3);
      fa.q[0] = *(const u32x4*)ap;
      fa.q[1] = *(const u32x4*)(ap + 16);
    } else {
#pragma unroll
      for (int e = 0; e < 16; ++e) {
        int ka = kk + ((e & 8) ? (16 + (half << 3) + (e & 7)) : ((half << 3) + e));
        fa.v[e] = A[(size_t)ka * lda + (row0 + r)];
      }
    }
    const u32x4* sp = (const u32x4*)&sB[buf][lane << 4];
    fb.q[0] = sp[0];                          // ds_load_b128
    fb.q[1] = sp[1];
    acc = __builtin_amdgcn_wmma_f32_16x16x32_bf16(
        false, fa.v, false, fb.v, (short)0, acc, false, false);
    // no trailing barrier: next iteration writes the other LDS buffer, and
    // the next same-buffer write is ordered behind the following barrier.
  }
#pragma unroll
  for (int i = 0; i < 8; ++i)
    Dm[(size_t)(row0 + (half << 3) + i) * ldd + col0 + r] = acc[i];
}

// ---- BT=1 GEMM (Gram: B = z^T), fully vectorized direct loads -------------
__global__ void k_gemm_nt(const __bf16* __restrict__ A, const __bf16* __restrict__ B,
                          float* __restrict__ Dm, int M, int N, int K,
                          int lda, int ldb, int ldd) {
  const int ntn = N >> 4;
  const int totalTiles = (M >> 4) * ntn;
  const int wave = (int)((blockIdx.x * blockDim.x + threadIdx.x) >> 5);
  if (wave >= totalTiles) return;  // wave-uniform: EXEC stays all-ones
  const int tm = wave / ntn, tn = wave % ntn;
  const int lane = threadIdx.x & 31;
  const int half = lane >> 4, r = lane & 15;
  const int row0 = tm << 4, col0 = tn << 4;

  f32x8 acc = {};
  for (int kk = 0; kk < K; kk += 64) {   // K multiple of 64
#pragma unroll
    for (int s = 0; s < 2; ++s) {
      const int k0 = kk + (s << 5);
      union { bf16x16 v; u32x4 q[2]; } fa, fb;
      const __bf16* ap = A + (size_t)(row0 + r) * lda + k0 + (half << 3);
      fa.q[0] = *(const u32x4*)ap;
      fa.q[1] = *(const u32x4*)(ap + 16);
      const __bf16* bp = B + (size_t)(col0 + r) * ldb + k0 + (half << 3);
      fb.q[0] = *(const u32x4*)bp;
      fb.q[1] = *(const u32x4*)(bp + 16);
      acc = __builtin_amdgcn_wmma_f32_16x16x32_bf16(
          false, fa.v, false, fb.v, (short)0, acc, false, false);
    }
  }
#pragma unroll
  for (int i = 0; i < 8; ++i)
    Dm[(size_t)(row0 + (half << 3) + i) * ldd + col0 + r] = acc[i];
}

// ------------------------------- host driver -------------------------------

extern "C" void kernel_launch(void* const* d_in, const int* in_sizes, int n_in,
                              void* d_out, int out_size, void* d_ws, size_t ws_size,
                              hipStream_t stream) {
  (void)n_in; (void)ws_size; (void)out_size;
  const float* C  = (const float*)d_in[0];
  const float* zA = (const float*)d_in[1];
  const float* zB = (const float*)d_in[2];
  const int NA = in_sizes[3];                 // mask_A length
  const int NB = in_sizes[4];                 // mask_B length
  const int D  = in_sizes[1] / NA;            // embedding dim
  const float fi = REG_M / (REG_M + REG_EPS);

  // -------- workspace layout. f32: 4*NA*NB + 5*NMAX*D + small.
  //          bf16: 2*NA*NB + (NA+NB+NMAX)*D.  (~93 MB total @ 2048/256)
  float* ws = (float*)d_ws;
  const size_t nn = (size_t)NA * NB;
  float* Km = ws;             // K, later A_adj scratch
  float* S1 = ws + nn;        // K^T, later M_A / Gram_B
  float* P  = ws + 2 * nn;    // transport plan
  float* S3 = ws + 3 * nn;    // Gram_A / M_B
  const int NMAX = NA > NB ? NA : NB;
  float* u   = ws + 4 * nn;   // NA
  float* v   = u + NA;        // NB
  float* sq  = v + NB;        // NMAX
  float* dn  = sq + NMAX;     // NMAX
  float* rr0 = dn + NMAX;     // D
  float* rr1 = rr0 + D;       // D
  float* pAp = rr1 + D;       // D
  const size_t nd = (size_t)NMAX * D;
  float* bary = pAp + D;
  float* Xs   = bary + nd;
  float* Rs   = Xs + nd;
  float* Dir  = Rs + nd;
  float* AP   = Dir + nd;
  // bf16 staging area (16B aligned: all preceding counts are multiples of 4)
  __bf16* P_bf   = (__bf16*)(AP + nd);          // nn
  __bf16* M_bf   = P_bf + nn;                   // nn
  __bf16* zA_bf  = M_bf + nn;                   // NA*D
  __bf16* zB_bf  = zA_bf + (size_t)NA * D;      // NB*D
  __bf16* Dir_bf = zB_bf + (size_t)NB * D;      // nd

  const int TPB = 256;
  const int EW_BLK = 2048;  // grid-stride elementwise blocks

  // ---- init output, build K / K^T, pack z to bf16 ----------------------
  k_fill<<<1, 64, 0, stream>>>((float*)d_out, 1, 0.0f);
  k_exp_build<<<EW_BLK, TPB, 0, stream>>>(C, Km, S1, NA, NB);
  k_cvt_bf16<<<EW_BLK, TPB, 0, stream>>>(zA, zA_bf, (long)NA * D);
  k_cvt_bf16<<<EW_BLK, TPB, 0, stream>>>(zB, zB_bf, (long)NB * D);

  // ---- unbalanced Sinkhorn (f32, L2-resident, wave-per-row) -------------
  k_fill<<<(NA + TPB - 1) / TPB, TPB, 0, stream>>>(u, NA, 1.0f);
  k_fill<<<(NB + TPB - 1) / TPB, TPB, 0, stream>>>(v, NB, 1.0f);
  for (int it = 0; it < N_SINK; ++it) {
    k_sink<<<(NA + 7) / 8, TPB, 0, stream>>>(Km, v, u, NA, NB, 1.0f / NA, fi);
    k_sink<<<(NB + 7) / 8, TPB, 0, stream>>>(S1, u, v, NB, NA, 1.0f / NB, fi);
  }
  k_build_P<<<EW_BLK, TPB, 0, stream>>>(Km, u, v, P, NA, NB);
  k_cvt_bf16<<<EW_BLK, TPB, 0, stream>>>(P, P_bf, (long)nn);

  // ---- per-side: bary -> kNN Laplacian -> CG solve -> loss --------------
  for (int side = 0; side < 2; ++side) {
    const float*  zs    = side == 0 ? zA : zB;
    const __bf16* zs_bf = side == 0 ? zA_bf : zB_bf;
    const __bf16* zt_bf = side == 0 ? zB_bf : zA_bf;
    const int ns = side == 0 ? NA : NB;
    float* Gm  = side == 0 ? S3 : S1;   // Gram scratch
    float* Mop = side == 0 ? S1 : S3;   // dense I + lam*L

    // bary = (plan @ zt) / (rowsum(plan) + eps);  plan = P (side0) / P^T (side1)
    if (side == 0) {
      k_rowsum<<<(NA + 7) / 8, TPB, 0, stream>>>(P, dn, NA, NB);
      k_gemm_sh<0><<<(NA / 128) * (D / 16), TPB, 0, stream>>>(
          P_bf, zt_bf, bary, NA, D, NB, NB, D, D);
    } else {
      k_colsum<<<(NB + 7) / 8, TPB, 0, stream>>>(P, dn, NA, NB);
      k_gemm_sh<1><<<(NB / 128) * (D / 16), TPB, 0, stream>>>(
          P_bf, zt_bf, bary, NB, D, NA, NB, D, D);
    }
    k_scale_rows<<<EW_BLK, TPB, 0, stream>>>(bary, dn, ns, D);

    // Gram (z z^T) -> kNN adjacency -> M = I + lam*(deg - max(A,A^T))
    k_rownormsq<<<(ns + 7) / 8, TPB, 0, stream>>>(zs, sq, ns, D);
    k_gemm_nt<<<(((ns / 16) * (ns / 16)) + 7) / 8, TPB, 0, stream>>>(
        zs_bf, zs_bf, Gm, ns, ns, D, D, D, ns);
    k_fill<<<EW_BLK, TPB, 0, stream>>>(Km, (long)ns * ns, 0.0f);
    int kk = K_NN < ns - 1 ? K_NN : ns - 1;
    k_knn<<<ns, TPB, 0, stream>>>(Gm, sq, Km, ns, kk);
    k_build_M<<<ns, TPB, 0, stream>>>(Km, Mop, ns, LAM_EIG);
    k_cvt_bf16<<<EW_BLK, TPB, 0, stream>>>(Mop, M_bf, (long)ns * ns);

    // CG: (I + lam*L) X = bary   (== V filt V^T bary)
    k_cg_init<<<EW_BLK, TPB, 0, stream>>>(Xs, Rs, Dir, bary, (long)ns * D);
    k_coldot<<<D, TPB, 0, stream>>>(Rs, Rs, rr0, ns, D);
    float* rc = rr0; float* rn = rr1;
    for (int it = 0; it < N_CG; ++it) {
      k_cvt_bf16<<<EW_BLK, TPB, 0, stream>>>(Dir, Dir_bf, (long)ns * D);
      k_gemm_sh<0><<<(ns / 128) * (D / 16), TPB, 0, stream>>>(
          M_bf, Dir_bf, AP, ns, D, ns, ns, D, D);
      k_coldot<<<D, TPB, 0, stream>>>(Dir, AP, pAp, ns, D);
      k_cg_xr<<<EW_BLK, TPB, 0, stream>>>(Xs, Rs, Dir, AP, rc, pAp, (long)ns * D, D);
      k_coldot<<<D, TPB, 0, stream>>>(Rs, Rs, rn, ns, D);
      k_cg_dir<<<EW_BLK, TPB, 0, stream>>>(Dir, Rs, rn, rc, (long)ns * D, D);
      float* t = rc; rc = rn; rn = t;
    }

    k_loss<<<256, TPB, 0, stream>>>(zs, Xs, (long)ns * D,
                                    1.0f / ((float)ns * (float)D), (float*)d_out);
  }
}
